// Feature_RNN_for_single_seg_prediction_27943057228467
// MI455X (gfx1250) — matrix operand order repaired and checked by reading, not compiled, not run on hardware
//
#include <hip/hip_runtime.h>
#include <hip/hip_bf16.h>
#include <math.h>

// ---------------------------------------------------------------------------
// CDNA5 (gfx1250) wave32 WMMA RNN implementation, round 4.
//  - bf16 weights in workspace -> B frags are raw global_load_b128
//  - split accumulators in the recurrence (2 independent WMMA chains / tile)
//  - async global->LDS staging for the embedding gather (ASYNCcnt path),
//    builtin prototype: (global v4i*, local v4i*, imm offset, imm cpol)
// ---------------------------------------------------------------------------

typedef __attribute__((ext_vector_type(16))) unsigned short v16u;
typedef __attribute__((ext_vector_type(16))) __bf16         v16bf;
typedef __attribute__((ext_vector_type(8)))  float          v8f;
typedef __attribute__((ext_vector_type(4)))  int            v4i;

typedef __attribute__((address_space(1))) v4i* as1_v4i_p;   // global
typedef __attribute__((address_space(3))) v4i* as3_v4i_p;   // LDS

#define HDIM 256
#define LDSW (HDIM + 8)   // ushort row stride, keeps rows 16B aligned

__device__ __forceinline__ unsigned short f2bf(float f) {
    __bf16 h = (__bf16)f;                       // hw cvt on gfx1250
    return __builtin_bit_cast(unsigned short, h);
}

__device__ __forceinline__ unsigned pk2bf(float lo, float hi) {
#if __has_builtin(__builtin_amdgcn_cvt_pk_bf16_f32)
    typedef __attribute__((ext_vector_type(2))) __bf16 v2bf;
    v2bf p = __builtin_amdgcn_cvt_pk_bf16_f32(lo, hi);
    return __builtin_bit_cast(unsigned, p);
#else
    return (unsigned)f2bf(lo) | ((unsigned)f2bf(hi) << 16);
#endif
}

__device__ __forceinline__ float fast_tanh(float x) {
#if __has_builtin(__builtin_amdgcn_tanhf)
    return __builtin_amdgcn_tanhf(x);           // v_tanh_f32 (TRANS32)
#elif __has_builtin(__builtin_amdgcn_tanh_f32)
    return __builtin_amdgcn_tanh_f32(x);
#else
    return tanhf(x);
#endif
}

__device__ __forceinline__ void wait_asynccnt0() {
#if __has_builtin(__builtin_amdgcn_s_wait_asynccnt)
    __builtin_amdgcn_s_wait_asynccnt(0);
#else
    asm volatile("s_wait_asynccnt 0x0" ::: "memory");
#endif
}

// A fragment (16x32 bf16) from an LDS tile, row-major, stride ldr (ushorts).
// ISA layout: lane m=lane%16, half hh=lane/16;
//   VGPR0..3 hold K = kbase + hh*8 + [0..7]      (contiguous 8 -> ds_load_b128)
//   VGPR4..7 hold K = kbase + 16 + hh*8 + [0..7] (contiguous 8 -> ds_load_b128)
__device__ __forceinline__ v16bf load_a_lds(const unsigned short* lds, int ldr,
                                            int m, int hh, int kbase) {
    const uint4* p0 = (const uint4*)(lds + m * ldr + kbase + hh * 8);
    const uint4* p1 = (const uint4*)(lds + m * ldr + kbase + 16 + hh * 8);
    v16u v;
    ((uint4*)&v)[0] = *p0;
    ((uint4*)&v)[1] = *p1;
    return __builtin_bit_cast(v16bf, v);
}

// B fragment (32x16 bf16) where logical B = W^T, W row-major [N, K] **bf16**.
// ISA layout: lane n=lane%16, half hh=lane/16;
//   K = kbase + hh*8 + [0..7], kbase + 16 + hh*8 + [0..7] -> two global_load_b128.
__device__ __forceinline__ v16bf load_b_bf(const unsigned short* W, int ldw,
                                           int n, int hh, int kbase) {
    const uint4* p0 = (const uint4*)(W + (size_t)n * ldw + kbase + hh * 8);
    const uint4* p1 = (const uint4*)(W + (size_t)n * ldw + kbase + 16 + hh * 8);
    v16u v;
    ((uint4*)&v)[0] = *p0;
    ((uint4*)&v)[1] = *p1;
    return __builtin_bit_cast(v16bf, v);
}

__device__ __forceinline__ v8f wmma_bf16(v16bf a, v16bf b, v8f c) {
    return __builtin_amdgcn_wmma_f32_16x16x32_bf16(false, a, false, b,
                                                   (short)0, c, false, false);
}

// ---------------------------------------------------------------------------
// Kernel 0: bulk f32 -> bf16 conversion (weights, embedding table).
// ---------------------------------------------------------------------------
__global__ __launch_bounds__(256) void cvt_bf16_kernel(
    const float* __restrict__ src, unsigned* __restrict__ dst, long npairs) {
    long i = (long)blockIdx.x * blockDim.x + threadIdx.x;
    long stride = (long)gridDim.x * blockDim.x;
    for (; i < npairs; i += stride)
        dst[i] = pk2bf(src[2 * i], src[2 * i + 1]);
}

// ---------------------------------------------------------------------------
// Kernel 1/3: tiled projection GEMM  Out[r, 0:256] = X[r, 0:K] @ W^T + bias
// If tokens != nullptr, X row r is gathered from bf16 table FeatBF[tok, :]
// (async global->LDS when available). Fully in-place safe (tile staged in LDS).
// Grid: R/16 blocks of 256 threads (8 waves); wave w owns columns 32w..32w+31.
// ---------------------------------------------------------------------------
__global__ __launch_bounds__(256) void proj_gemm_kernel(
    const float* __restrict__ X, const int* __restrict__ tokens,
    const unsigned short* __restrict__ FeatBF,
    const unsigned short* __restrict__ Wbf,
    const float* __restrict__ b0, const float* __restrict__ b1,
    float* __restrict__ Out, int K) {
    __shared__ unsigned short xb[16][LDSW];

    const int wave = threadIdx.x >> 5, lane = threadIdx.x & 31;
    const int n = lane & 15, hh = lane >> 4;
    const size_t rb = (size_t)blockIdx.x * 16;

    // stage input tile (bf16) into LDS
    if (tokens) {
#if __has_builtin(__builtin_amdgcn_global_load_async_to_lds_b128)
        // async path: 16B per lane, ASYNCcnt-tracked; K assumed multiple of 8
        {
            const int nchunk = K >> 3;                 // 16B chunks per row
            for (int i = threadIdx.x; i < 16 * nchunk; i += 256) {
                int r = i / nchunk, c = i - r * nchunk;
                int tok = tokens[rb + r];
                const unsigned short* g = FeatBF + (size_t)tok * K + c * 8;
                __builtin_amdgcn_global_load_async_to_lds_b128(
                    (as1_v4i_p)(size_t)g,
                    (as3_v4i_p)(size_t)(unsigned)(size_t)&xb[r][c * 8],
                    0, 0);
            }
            wait_asynccnt0();
        }
#else
        for (int i = threadIdx.x; i < 16 * K; i += 256) {
            int r = i / K, k = i - r * K;
            int tok = tokens[rb + r];
            xb[r][k] = FeatBF[(size_t)tok * K + k];
        }
#endif
    } else {
        // f32 source: hardware packed cvt, 2 elements per iteration
        for (int i = threadIdx.x; i < (16 * K) / 2; i += 256) {
            int r = i / (K / 2), k2 = i - r * (K / 2);
            const float* p = X + (rb + r) * (size_t)K + 2 * k2;
            *(unsigned*)&xb[r][2 * k2] = pk2bf(p[0], p[1]);
        }
    }
    __syncthreads();

    const int nkt = K >> 5;
#pragma unroll
    for (int j = 0; j < 2; ++j) {
        const int cb = wave * 32 + j * 16;
        const float bias = b0[cb + n] + b1[cb + n];
        v8f acc;
#pragma unroll
        for (int r = 0; r < 8; ++r) acc[r] = bias;
        for (int kt = 0; kt < nkt; ++kt) {
            v16bf a  = load_a_lds(&xb[0][0], LDSW, lane & 15, hh, kt * 32);
            v16bf bw = load_b_bf(Wbf, K, cb + n, hh, kt * 32);
            acc = wmma_bf16(a, bw, acc);
        }
#pragma unroll
        for (int r = 0; r < 8; ++r) {
            int m = r + 8 * hh;
            Out[(rb + m) * (size_t)HDIM + cb + n] = acc[r];
        }
    }
}

// ---------------------------------------------------------------------------
// Kernel 2/4: sequential tanh recurrence, in place on seq[B,S,H]:
//   h_t = tanh(seq[:,t,:] + h_{t-1} @ W_hh^T);  seq[:,t,:] <- h_t
// One WG per 16 batch rows; W_hh fragments pinned in VGPRs (16 x v16bf);
// h tile in LDS as bf16; 2 accumulators per tile to halve the dependent
// WMMA chain on the latency-critical path.
// ---------------------------------------------------------------------------
__global__ __launch_bounds__(256) void rnn_recur_kernel(
    float* __restrict__ seq, const unsigned short* __restrict__ Whhbf, int S) {
    __shared__ unsigned short hbuf[16][LDSW];

    const int wave = threadIdx.x >> 5, lane = threadIdx.x & 31;
    const int n = lane & 15, hh = lane >> 4;
    const size_t rb = (size_t)blockIdx.x * 16;

    // W_hh^T fragments resident in registers for the whole time loop
    v16bf wfr[2][8];
#pragma unroll
    for (int j = 0; j < 2; ++j) {
        const int ng = wave * 32 + j * 16 + n;
#pragma unroll
        for (int kt = 0; kt < 8; ++kt)
            wfr[j][kt] = load_b_bf(Whhbf, HDIM, ng, hh, kt * 32);
    }

    // h_0 = 0
    for (int i = threadIdx.x; i < 16 * LDSW; i += 256)
        (&hbuf[0][0])[i] = 0;
    __syncthreads();

    for (int t = 0; t < S; ++t) {
        // prefetch next timestep's xW slice for this wave's columns
        if (t + 1 < S)
            __builtin_prefetch(
                &seq[((rb + 8 * hh) * S + (t + 1)) * (size_t)HDIM + wave * 32 + n],
                0, 1);

        // A fragments of h_{t-1} from LDS
        v16bf afr[8];
#pragma unroll
        for (int kt = 0; kt < 8; ++kt)
            afr[kt] = load_a_lds(&hbuf[0][0], LDSW, lane & 15, hh, kt * 32);

        v8f acc[2][2];
#pragma unroll
        for (int j = 0; j < 2; ++j) {
            const int cb = wave * 32 + j * 16;
#pragma unroll
            for (int r = 0; r < 8; ++r) {
                int m = r + 8 * hh;
                acc[j][0][r] = seq[((rb + m) * S + t) * (size_t)HDIM + cb + n];
                acc[j][1][r] = 0.0f;
            }
            // two independent K-chains (even / odd tiles)
#pragma unroll
            for (int kt = 0; kt < 8; ++kt)
                acc[j][kt & 1] = wmma_bf16(afr[kt], wfr[j][kt], acc[j][kt & 1]);
        }
        __syncthreads();  // all h_{t-1} reads complete before overwrite

#pragma unroll
        for (int j = 0; j < 2; ++j) {
            const int cb = wave * 32 + j * 16;
#pragma unroll
            for (int r = 0; r < 8; ++r) {
                int m = r + 8 * hh;
                float hv = fast_tanh(acc[j][0][r] + acc[j][1][r]);
                seq[((rb + m) * S + t) * (size_t)HDIM + cb + n] = hv;
                hbuf[m][cb + n] = f2bf(hv);
            }
        }
        __syncthreads();  // h_t visible before next step's reads
    }
}

// ---------------------------------------------------------------------------
// Kernel 5: logits = h_last[B,H] @ W_out^T[V,H] + b_out  -> out[B,V]
// Grid: (ceil(V/256), B/16); wave w owns N-tiles blockIdx.x*16 + 2w, +2w+1.
// ---------------------------------------------------------------------------
__global__ __launch_bounds__(256) void out_gemm_kernel(
    const float* __restrict__ seq, const unsigned short* __restrict__ Woutbf,
    const float* __restrict__ b_out, float* __restrict__ out,
    int S, int V) {
    __shared__ unsigned short xb[16][LDSW];

    const int wave = threadIdx.x >> 5, lane = threadIdx.x & 31;
    const int n = lane & 15, hh = lane >> 4;
    const size_t rb = (size_t)blockIdx.y * 16;

    // stage h_last tile (t = S-1) as bf16, packed cvt
    for (int i = threadIdx.x; i < (16 * HDIM) / 2; i += 256) {
        int r = i >> 7, k2 = i & 127;
        const float* p = seq + ((rb + r) * S + (S - 1)) * (size_t)HDIM + 2 * k2;
        *(unsigned*)&xb[r][2 * k2] = pk2bf(p[0], p[1]);
    }
    __syncthreads();

#pragma unroll
    for (int j = 0; j < 2; ++j) {
        const int ntile = blockIdx.x * 16 + wave * 2 + j;
        const int ng = ntile * 16 + n;
        const int ngc = ng < V ? ng : V - 1;  // clamp loads, keep EXEC full
        v8f acc;
#pragma unroll
        for (int r = 0; r < 8; ++r) acc[r] = 0.0f;
#pragma unroll
        for (int kt = 0; kt < 8; ++kt) {
            v16bf a  = load_a_lds(&xb[0][0], LDSW, lane & 15, hh, kt * 32);
            v16bf bw = load_b_bf(Woutbf, HDIM, ngc, hh, kt * 32);
            acc = wmma_bf16(a, bw, acc);
        }
        const float bias = b_out[ngc];
        if (ng < V) {
#pragma unroll
            for (int r = 0; r < 8; ++r) {
                int m = r + 8 * hh;
                out[(rb + m) * (size_t)V + ng] = acc[r] + bias;
            }
        }
    }
}

// ---------------------------------------------------------------------------
extern "C" void kernel_launch(void* const* d_in, const int* in_sizes, int n_in,
                              void* d_out, int out_size, void* d_ws, size_t ws_size,
                              hipStream_t stream) {
    (void)in_sizes; (void)n_in; (void)out_size; (void)ws_size;
    const int B = 256, S = 512, V = 50000, F = 64;

    const int*   batch    = (const int*)d_in[0];
    const float* features = (const float*)d_in[1];
    const float* W_ih0    = (const float*)d_in[2];
    const float* W_hh0    = (const float*)d_in[3];
    const float* b_ih0    = (const float*)d_in[4];
    const float* b_hh0    = (const float*)d_in[5];
    const float* W_ih1    = (const float*)d_in[6];
    const float* W_hh1    = (const float*)d_in[7];
    const float* b_ih1    = (const float*)d_in[8];
    const float* b_hh1    = (const float*)d_in[9];
    const float* W_out    = (const float*)d_in[10];
    const float* b_out    = (const float*)d_in[11];
    float*       out      = (float*)d_out;

    // ---- workspace layout ----
    char* ws = (char*)d_ws;
    float* seq = (float*)ws;                               // [B*S, 256] f32
    size_t off = (size_t)B * S * HDIM * sizeof(float);     // 134 217 728
    unsigned short* wih0bf = (unsigned short*)(ws + off);  off += (size_t)HDIM * F * 2;
    unsigned short* whh0bf = (unsigned short*)(ws + off);  off += (size_t)HDIM * HDIM * 2;
    unsigned short* wih1bf = (unsigned short*)(ws + off);  off += (size_t)HDIM * HDIM * 2;
    unsigned short* whh1bf = (unsigned short*)(ws + off);  off += (size_t)HDIM * HDIM * 2;
    unsigned short* woutbf = (unsigned short*)(ws + off);  off += (size_t)V * HDIM * 2;
    unsigned short* featbf = (unsigned short*)(ws + off);  off += (size_t)V * F * 2;

    // ---- weight / table conversion to bf16 ----
    cvt_bf16_kernel<<<64,  256, 0, stream>>>(W_ih0,    (unsigned*)wih0bf, (long)HDIM * F / 2);
    cvt_bf16_kernel<<<64,  256, 0, stream>>>(W_hh0,    (unsigned*)whh0bf, (long)HDIM * HDIM / 2);
    cvt_bf16_kernel<<<64,  256, 0, stream>>>(W_ih1,    (unsigned*)wih1bf, (long)HDIM * HDIM / 2);
    cvt_bf16_kernel<<<64,  256, 0, stream>>>(W_hh1,    (unsigned*)whh1bf, (long)HDIM * HDIM / 2);
    cvt_bf16_kernel<<<512, 256, 0, stream>>>(W_out,    (unsigned*)woutbf, (long)V * HDIM / 2);
    cvt_bf16_kernel<<<512, 256, 0, stream>>>(features, (unsigned*)featbf, (long)V * F / 2);

    const int R = B * S;  // 131072 flattened rows

    // Layer 0: gather + input projection -> seq = emb @ W_ih0^T + (b_ih0+b_hh0)
    proj_gemm_kernel<<<R / 16, 256, 0, stream>>>(
        nullptr, batch, featbf, wih0bf, b_ih0, b_hh0, seq, F);
    // Layer 0: recurrence (in place)
    rnn_recur_kernel<<<B / 16, 256, 0, stream>>>(seq, whh0bf, S);
    // Layer 1: input projection (in place) -> seq = h0 @ W_ih1^T + (b_ih1+b_hh1)
    proj_gemm_kernel<<<R / 16, 256, 0, stream>>>(
        seq, nullptr, nullptr, wih1bf, b_ih1, b_hh1, seq, HDIM);
    // Layer 1: recurrence (in place)
    rnn_recur_kernel<<<B / 16, 256, 0, stream>>>(seq, whh1bf, S);
    // Output projection from h1[:, S-1, :]
    dim3 og((V + 255) / 256, B / 16);
    out_gemm_kernel<<<og, 256, 0, stream>>>(seq, woutbf, b_out, out, S, V);
}
